// BitLinear158_17703855194327
// MI455X (gfx1250) — 8-bit matrix operand order repaired and checked by reading, compiled, not run on hardware
//
#include <hip/hip_runtime.h>
#include <stdint.h>

#define TOKENS 4096
#define IN_F   8192
#define OUT_F  8192

typedef __attribute__((ext_vector_type(8))) int v8i;

__global__ __launch_bounds__(256)
void bitlinear_quant_kernel(const float* __restrict__ x,
                            signed char* __restrict__ xq,
                            float* __restrict__ inv_s) {
    __shared__ float red[256];
    const int row = blockIdx.x;
    const int t   = threadIdx.x;
    const float* xr = x + (size_t)row * IN_F;

    float m = 0.0f;
    for (int i = t; i < IN_F; i += 256)
        m = fmaxf(m, fabsf(xr[i]));
    red[t] = m;
    __syncthreads();
    for (int off = 128; off > 0; off >>= 1) {
        if (t < off) red[t] = fmaxf(red[t], red[t + off]);
        __syncthreads();
    }
    const float mx = fmaxf(red[0], 1e-5f);
    const float s  = 127.0f / mx;
    if (t == 0) inv_s[row] = mx / 127.0f;

    uint32_t* xqr = (uint32_t*)(xq + (size_t)row * IN_F);
    #pragma unroll
    for (int j = 0; j < IN_F / 1024; ++j) {
        const int idx = j * 1024 + t * 4;
        float4 v = *(const float4*)(xr + idx);
        int q0 = max(-128, min(127, __float2int_rn(v.x * s)));
        int q1 = max(-128, min(127, __float2int_rn(v.y * s)));
        int q2 = max(-128, min(127, __float2int_rn(v.z * s)));
        int q3 = max(-128, min(127, __float2int_rn(v.w * s)));
        xqr[idx >> 2] = (uint32_t)(q0 & 0xFF) | ((uint32_t)(q1 & 0xFF) << 8) |
                        ((uint32_t)(q2 & 0xFF) << 16) | ((uint32_t)(q3 & 0xFF) << 24);
    }
}

__global__ __launch_bounds__(256)
void bitlinear_packw_kernel(const int* __restrict__ w,
                            signed char* __restrict__ wq) {
    const size_t base = ((size_t)blockIdx.x * 256 + threadIdx.x) * 16;
    const int4* p = (const int4*)(w + base);
    int4 a = p[0], b = p[1], c = p[2], d = p[3];
    int4 o;
    o.x = (a.x & 0xFF) | ((a.y & 0xFF) << 8) | ((a.z & 0xFF) << 16) | ((a.w & 0xFF) << 24);
    o.y = (b.x & 0xFF) | ((b.y & 0xFF) << 8) | ((b.z & 0xFF) << 16) | ((b.w & 0xFF) << 24);
    o.z = (c.x & 0xFF) | ((c.y & 0xFF) << 8) | ((c.z & 0xFF) << 16) | ((c.w & 0xFF) << 24);
    o.w = (d.x & 0xFF) | ((d.y & 0xFF) << 8) | ((d.z & 0xFF) << 16) | ((d.w & 0xFF) << 24);
    *(int4*)(wq + base) = o;
}

__device__ __forceinline__ v8i load_a_frag(const signed char* base) {
    int2 d0 = *(const int2*)(base +  0);
    int2 d1 = *(const int2*)(base +  8);
    int2 d2 = *(const int2*)(base + 32);
    int2 d3 = *(const int2*)(base + 40);
    v8i a;
    a[0] = d0.x; a[1] = d0.y; a[2] = d1.x; a[3] = d1.y;
    a[4] = d2.x; a[5] = d2.y; a[6] = d3.x; a[7] = d3.y;
    return a;
}

__device__ __forceinline__ v8i load_b_frag(const signed char* base) {
    int4 d0 = *(const int4*)(base +  0);
    int4 d1 = *(const int4*)(base + 32);
    v8i b;
    b[0] = d0.x; b[1] = d0.y; b[2] = d0.z; b[3] = d0.w;
    b[4] = d1.x; b[5] = d1.y; b[6] = d1.z; b[7] = d1.w;
    return b;
}

__global__ __launch_bounds__(256)
void bitlinear_gemm_kernel(const signed char* __restrict__ xq,
                           const signed char* __restrict__ wq,
                           const float* __restrict__ inv_s,
                           const float* __restrict__ weight_scale,
                           const float* __restrict__ bias,
                           float* __restrict__ out) {
    const int lane = threadIdx.x & 31;
    const int wave = threadIdx.x >> 5;
    const int wm   = wave & 3;
    const int wn   = wave >> 2;
    const int m0   = blockIdx.y * 128 + wm * 32;
    const int n0   = blockIdx.x * 128 + wn * 64;
    const int lr   = lane & 15;
    const int hi   = lane >> 4;

    const signed char* aptr[2];
    #pragma unroll
    for (int m = 0; m < 2; ++m)
        aptr[m] = xq + (size_t)(m0 + m * 16 + lr) * IN_F + hi * 16;
    const signed char* bptr[4];
    #pragma unroll
    for (int n = 0; n < 4; ++n)
        bptr[n] = wq + (size_t)(n0 + n * 16 + lr) * IN_F + hi * 16;

    v8i acc[2][4] = {};

    #pragma unroll 2
    for (int k = 0; k < IN_F; k += 64) {
        v8i a[2], b[4];
        #pragma unroll
        for (int m = 0; m < 2; ++m) a[m] = load_a_frag(aptr[m] + k);
        #pragma unroll
        for (int n = 0; n < 4; ++n) b[n] = load_b_frag(bptr[n] + k);
        #pragma unroll
        for (int m = 0; m < 2; ++m)
            #pragma unroll
            for (int n = 0; n < 4; ++n)
                acc[m][n] = __builtin_amdgcn_wmma_i32_16x16x64_iu8(
                     true, a[m],  false, b[n],
                    acc[m][n],  false,  false);
    }

    const float wsinv = 1.0f / weight_scale[0];
    float rs[2][8];
    #pragma unroll
    for (int m = 0; m < 2; ++m) {
        const int mb = m0 + m * 16 + hi * 8;
        #pragma unroll
        for (int r = 0; r < 8; ++r) rs[m][r] = inv_s[mb + r] * wsinv;
    }

    #pragma unroll
    for (int n = 0; n < 4; ++n) {
        const int nc = n0 + n * 16 + lr;
        const float bv = bias[nc];
        #pragma unroll
        for (int m = 0; m < 2; ++m) {
            const int mb = m0 + m * 16 + hi * 8;
            float* op = out + (size_t)mb * OUT_F + nc;
            #pragma unroll
            for (int r = 0; r < 8; ++r)
                op[(size_t)r * OUT_F] = (float)acc[m][n][r] * rs[m][r] + bv;
        }
    }
}

extern "C" void kernel_launch(void* const* d_in, const int* in_sizes, int n_in,
                              void* d_out, int out_size, void* d_ws, size_t ws_size,
                              hipStream_t stream) {
    (void)in_sizes; (void)n_in; (void)out_size; (void)ws_size;
    const float* x      = (const float*)d_in[0];
    const int*   w      = (const int*)d_in[1];
    const float* wscale = (const float*)d_in[2];
    const float* bias   = (const float*)d_in[3];
    float*       out    = (float*)d_out;

    signed char* xq    = (signed char*)d_ws;
    signed char* wq    = xq + (size_t)TOKENS * IN_F;
    float*       inv_s = (float*)(wq + (size_t)OUT_F * IN_F);

    bitlinear_quant_kernel<<<TOKENS, 256, 0, stream>>>(x, xq, inv_s);

    const int pack_blocks = (int)(((size_t)OUT_F * IN_F) / 16 / 256);
    bitlinear_packw_kernel<<<pack_blocks, 256, 0, stream>>>(w, wq);

    dim3 grid(OUT_F / 128, TOKENS / 128);
    bitlinear_gemm_kernel<<<grid, 256, 0, stream>>>(xq, wq, inv_s, wscale, bias, out);
}
